// VectorQuantizer_20478404068042
// MI455X (gfx1250) — matrix operand order repaired and checked by reading, not compiled, not run on hardware
//
#include <hip/hip_runtime.h>
#include <hip/hip_bf16.h>

// ---------------- problem constants ----------------
#define NROWS   8192          // 8*1024 input vectors
#define DDIM    512
#define KCODES  16384
#define CHUNK   32            // codes staged in LDS per iteration
#define NSPLIT  8             // codebook slices across blockIdx.y
#define NCHUNKS ((KCODES / NSPLIT) / CHUNK)   // 64
#define CVEC    (CHUNK * DDIM / 8)            // uint4 per chunk = 2048
#define QUANT_ELEMS (NROWS * DDIM)       // 4194304
#define LOSS_OFF    QUANT_ELEMS          // d_out[4194304] = loss
#define IDX_OFF     (QUANT_ELEMS + 1)    // d_out[4194305..] = indices (as float)

// workspace layout (bytes)
#define XBF_OFF    0u
#define CBF_OFF    (NROWS * DDIM * 2u)                       //  8388608
#define CNORM_OFF  (CBF_OFF + KCODES * DDIM * 2u)            // 25165824
#define PVAL_OFF   (CNORM_OFF + KCODES * 4u)                 // 25231360
#define PIDX_OFF   (PVAL_OFF + NSPLIT * NROWS * 4u)          // 25493504

typedef __attribute__((ext_vector_type(16))) __bf16 v16bf;
typedef __attribute__((ext_vector_type(8)))  float  v8f;
typedef __attribute__((ext_vector_type(4)))  int    v4i;

typedef __attribute__((address_space(1))) v4i* gas_v4i_ptr;   // global
typedef __attribute__((address_space(3))) v4i* las_v4i_ptr;   // LDS

union Frag {
    v16bf v;
    uint4 q[2];   // 32 bytes = 16 bf16 elements
};

#if defined(__has_builtin)
#if __has_builtin(__builtin_amdgcn_global_load_async_to_lds_b128)
#define HAVE_ASYNC_LDS 1
#endif
#if __has_builtin(__builtin_amdgcn_sched_group_barrier)
#define HAVE_SCHED_GB 1
#endif
#endif

// ---------------- kernel 1: f32 -> bf16 conversion ----------------
__global__ __launch_bounds__(256) void vq_convert(const float* __restrict__ x,
                                                  const float* __restrict__ cb,
                                                  unsigned short* __restrict__ xbf,
                                                  unsigned short* __restrict__ cbf,
                                                  float* __restrict__ lossPtr) {
    size_t i = (size_t)blockIdx.x * 256 + threadIdx.x;
    if (i < (size_t)KCODES * DDIM) {
        __hip_bfloat16 h = __float2bfloat16(cb[i]);
        cbf[i] = *reinterpret_cast<unsigned short*>(&h);
    }
    if (i < (size_t)NROWS * DDIM) {
        __hip_bfloat16 h = __float2bfloat16(x[i]);
        xbf[i] = *reinterpret_cast<unsigned short*>(&h);
    }
    if (i == 0) *lossPtr = 0.0f;
}

// ---------------- kernel 2: codebook squared norms ----------------
__global__ __launch_bounds__(256) void vq_norms(const float* __restrict__ cb,
                                                float* __restrict__ cnorm) {
    int c = blockIdx.x * 256 + threadIdx.x;
    if (c < KCODES) {
        const float* row = cb + (size_t)c * DDIM;
        float s = 0.0f;
        #pragma unroll 8
        for (int j = 0; j < DDIM; ++j) { float v = row[j]; s += v * v; }
        cnorm[c] = s;
    }
}

// ---------------- kernel 3: bf16 WMMA score GEMM + running argmin ----------------
// grid = (32, NSPLIT) x 256 threads (8 waves).
// Wave w owns 32 rows (two 16-row A-tiles) at blockIdx.x*256 + w*32;
// slice blockIdx.y owns codes [y*2048, (y+1)*2048).
// LDS chunk is ping-pong double-buffered via async global->LDS DMA so the
// copy of chunk i+1 overlaps WMMA on chunk i. Inside a chunk, two sequential
// N-tile passes; each pass runs 2 WMMA chains sharing each B fragment and a
// sched_group_barrier-enforced DS->WMMA software pipeline (distance 2).
__global__ __launch_bounds__(256) void vq_gemm_argmin(const unsigned short* __restrict__ Xbf,
                                                      const unsigned short* __restrict__ Cbf,
                                                      const float* __restrict__ cnorm,
                                                      float* __restrict__ pVal,
                                                      int* __restrict__ pIdx) {
    __shared__ uint4 ldsC[2][CVEC];   // 2 x 32 KB ping-pong

    const int tid     = threadIdx.x;
    const int lane    = tid & 31;
    const int wave    = tid >> 5;
    const int half    = lane >> 4;        // 0: lanes 0-15, 1: lanes 16-31
    const int colLane = lane & 15;
    const int rowBase = blockIdx.x * 256 + wave * 32;
    const int cstart  = blockIdx.y * (KCODES / NSPLIT);

    const uint4* gX = reinterpret_cast<const uint4*>(Xbf);
    const uint4* gC = reinterpret_cast<const uint4*>(Cbf);

    // A fragments for two 16-row tiles, full D=512, resident in registers.
    // 16-bit A 16x32 layout: lanes<16 hold K[k0..k0+7] then K[k0+16..k0+23];
    // lanes>=16 hold K[k0+8..k0+15] then K[k0+24..k0+31].
    Frag fa[2][16];
    #pragma unroll
    for (int t = 0; t < 2; ++t) {
        const size_t xbase = (size_t)(rowBase + t * 16 + colLane) * DDIM;
        #pragma unroll
        for (int s = 0; s < 16; ++s) {
            const int k0 = s * 32;
            fa[t][s].q[0] = gX[(xbase + k0 +      half * 8) >> 3];
            fa[t][s].q[1] = gX[(xbase + k0 + 16 + half * 8) >> 3];
        }
    }

    float bv[2][8];
    int   bi[2][8];
    #pragma unroll
    for (int t = 0; t < 2; ++t)
        #pragma unroll
        for (int r = 0; r < 8; ++r) { bv[t][r] = 3.0e38f; bi[t][r] = 0; }

    // issue async copy of one chunk into an LDS buffer (8 b128 per wave-lane)
    auto issue_chunk = [&](int cbase, int buf) {
        #pragma unroll
        for (int e = tid; e < CVEC; e += 256) {
#if defined(HAVE_ASYNC_LDS)
            __builtin_amdgcn_global_load_async_to_lds_b128(
                (gas_v4i_ptr)(size_t)&gC[(size_t)cbase * (DDIM / 8) + e],
                (las_v4i_ptr)(size_t)&ldsC[buf][e],
                0, 0);
#else
            ldsC[buf][e] = gC[(size_t)cbase * (DDIM / 8) + e];
#endif
        }
    };

    issue_chunk(cstart, 0);   // prologue: chunk 0 -> buffer 0

    for (int ci = 0; ci < NCHUNKS; ++ci) {
        const int cbase = cstart + ci * CHUNK;
        const int buf   = ci & 1;

        if (ci + 1 < NCHUNKS) {
            issue_chunk(cbase + CHUNK, buf ^ 1);  // overlap DMA with compute
#if defined(HAVE_ASYNC_LDS)
#if __has_builtin(__builtin_amdgcn_s_wait_asynccnt)
            __builtin_amdgcn_s_wait_asynccnt(8);  // in-order: chunk ci complete
#else
            asm volatile("s_wait_asynccnt 0x8" ::: "memory");
#endif
#endif
        } else {
#if defined(HAVE_ASYNC_LDS)
#if __has_builtin(__builtin_amdgcn_s_wait_asynccnt)
            __builtin_amdgcn_s_wait_asynccnt(0);
#else
            asm volatile("s_wait_asynccnt 0x0" ::: "memory");
#endif
#endif
        }
        __syncthreads();   // every wave's portion of chunk ci is visible

        // two sequential N-tile passes on ldsC[buf]
        #pragma unroll
        for (int nt = 0; nt < 2; ++nt) {
            const int ln    = nt * 16 + colLane;        // B column (local code)
            const int ebase = (ln * DDIM + half * 8) >> 3;
            const uint4* lc = ldsC[buf];

            v8f acc0 = {}, acc1 = {};
            Frag fb[2];
            fb[0].q[0] = lc[ebase];
            fb[0].q[1] = lc[ebase + 2];
            #pragma unroll
            for (int s = 0; s < 16; ++s) {
                const int cur = s & 1;
                const int nxt = cur ^ 1;
                if (s < 15) {
                    const int e = ebase + (s + 1) * 4;   // 32 bf16 = 4 uint4 per kstep
                    fb[nxt].q[0] = lc[e];
                    fb[nxt].q[1] = lc[e + 2];
                }
                acc0 = __builtin_amdgcn_wmma_f32_16x16x32_bf16(
                           false, fa[0][s].v, false, fb[cur].v, (short)0, acc0, false, false);
                acc1 = __builtin_amdgcn_wmma_f32_16x16x32_bf16(
                           false, fa[1][s].v, false, fb[cur].v, (short)0, acc1, false, false);
            }
#if defined(HAVE_SCHED_GB)
            // shape this pass: 3 B-pairs in flight, then [2 WMMA | 2 DS-read]
            // steady state, tail of 6 WMMAs. 32 DS reads + 32 WMMAs per pass.
            __builtin_amdgcn_sched_group_barrier(0x100, 6, 0);   // DS read x6
            #pragma unroll
            for (int g = 0; g < 13; ++g) {
                __builtin_amdgcn_sched_group_barrier(0x008, 2, 0);  // WMMA x2
                __builtin_amdgcn_sched_group_barrier(0x100, 2, 0);  // DS read x2
            }
            __builtin_amdgcn_sched_group_barrier(0x008, 6, 0);   // WMMA x6
#endif

            const float cn   = cnorm[cbase + ln];   // global (vmem), not DS
            const int   code = cbase + ln;
            #pragma unroll
            for (int r = 0; r < 8; ++r) {
                const float s0 = cn - 2.0f * acc0[r];   // ||c||^2 - 2 x.c
                if (s0 < bv[0][r]) { bv[0][r] = s0; bi[0][r] = code; }
                const float s1 = cn - 2.0f * acc1[r];
                if (s1 < bv[1][r]) { bv[1][r] = s1; bi[1][r] = code; }
            }
        }

        __syncthreads();   // done reading ldsC[buf]; safe to overwrite next iter
    }

    // reduce min/argmin across the 16 lanes of each half (xor<=8 stays in-half)
    #pragma unroll
    for (int off = 8; off >= 1; off >>= 1) {
        #pragma unroll
        for (int t = 0; t < 2; ++t) {
            #pragma unroll
            for (int r = 0; r < 8; ++r) {
                const float ov = __shfl_xor(bv[t][r], off, 32);
                const int   oi = __shfl_xor(bi[t][r], off, 32);
                if (ov < bv[t][r] || (ov == bv[t][r] && oi < bi[t][r])) {
                    bv[t][r] = ov; bi[t][r] = oi;
                }
            }
        }
    }
    if (colLane == 0) {
        const int slice = blockIdx.y;
        #pragma unroll
        for (int t = 0; t < 2; ++t) {
            #pragma unroll
            for (int r = 0; r < 8; ++r) {
                const int row = rowBase + t * 16 + half * 8 + r;  // C/D row m = r + 8*half
                pVal[slice * NROWS + row] = bv[t][r];
                pIdx[slice * NROWS + row] = bi[t][r];
            }
        }
    }
}

// ---------------- kernel 4: merge slices + gather + outputs + loss ----------------
// one block per row
__global__ __launch_bounds__(256) void vq_finalize(const float* __restrict__ x,
                                                   const float* __restrict__ cb,
                                                   const float* __restrict__ pVal,
                                                   const int* __restrict__ pIdx,
                                                   float* __restrict__ out) {
    const int row = blockIdx.x;

    __shared__ int sIdx;
    if (threadIdx.x == 0) {
        float bestV = pVal[row];
        int   bestI = pIdx[row];
        #pragma unroll
        for (int p = 1; p < NSPLIT; ++p) {
            const float v = pVal[p * NROWS + row];
            const int   i = pIdx[p * NROWS + row];
            if (v < bestV || (v == bestV && i < bestI)) { bestV = v; bestI = i; }
        }
        sIdx = bestI;
    }
    __syncthreads();
    const int idx = sIdx;

    const float* crow = cb + (size_t)idx * DDIM;
    const float* xrow = x  + (size_t)row * DDIM;
    float*       orow = out + (size_t)row * DDIM;

    float local = 0.0f;
    for (int j = threadIdx.x; j < DDIM; j += 256) {
        const float q  = crow[j];
        const float xi = xrow[j];
        orow[j] = q;                      // straight-through value == quantized
        const float d = q - xi;
        local += d * d;
    }
    #pragma unroll
    for (int off = 16; off >= 1; off >>= 1)
        local += __shfl_xor(local, off, 32);

    __shared__ float part[8];
    if ((threadIdx.x & 31) == 0) part[threadIdx.x >> 5] = local;
    __syncthreads();
    if (threadIdx.x == 0) {
        float s = 0.0f;
        #pragma unroll
        for (int w = 0; w < 8; ++w) s += part[w];
        // loss = q_latent + 0.25*e_latent = 1.25 * mean((q-x)^2)
        atomicAdd(out + LOSS_OFF, s * (1.25f / (float)QUANT_ELEMS));
        out[IDX_OFF + row] = (float)idx;
    }
}

// ---------------- launcher ----------------
extern "C" void kernel_launch(void* const* d_in, const int* in_sizes, int n_in,
                              void* d_out, int out_size, void* d_ws, size_t ws_size,
                              hipStream_t stream) {
    const float* x  = (const float*)d_in[0];   // (8,1024,512) f32
    const float* cb = (const float*)d_in[1];   // (16384,512)  f32
    float* out = (float*)d_out;

    char* ws = (char*)d_ws;
    unsigned short* xbf   = (unsigned short*)(ws + XBF_OFF);
    unsigned short* cbf   = (unsigned short*)(ws + CBF_OFF);
    float*          cnorm = (float*)(ws + CNORM_OFF);
    float*          pval  = (float*)(ws + PVAL_OFF);
    int*            pidx  = (int*)(ws + PIDX_OFF);

    // 1) convert to bf16 + zero loss accumulator
    vq_convert<<<(KCODES * DDIM) / 256, 256, 0, stream>>>(x, cb, xbf, cbf, out + LOSS_OFF);
    // 2) codebook norms
    vq_norms<<<KCODES / 256, 256, 0, stream>>>(cb, cnorm);
    // 3) WMMA score GEMM + per-slice argmin: 32x8 WGs of 8 waves (32 rows/wave)
    dim3 g3(NROWS / 256, NSPLIT, 1);
    vq_gemm_argmin<<<g3, 256, 0, stream>>>(xbf, cbf, cnorm, pval, pidx);
    // 4) merge slices, gather quantized rows, loss, indices
    vq_finalize<<<NROWS, 256, 0, stream>>>(x, cb, pval, pidx, out);
}
